// GroupAttentionLayer_65343632441905
// MI455X (gfx1250) — matrix operand order repaired and checked by reading, not compiled, not run on hardware
//
#include <hip/hip_runtime.h>
#include <hip/hip_bf16.h>
#include <stdint.h>

#define ALPHA  0.3f
#define BN_EPS 0.001f

typedef __attribute__((ext_vector_type(16))) __bf16 v16bf;
typedef __attribute__((ext_vector_type(8)))  __bf16 v8bf;
typedef __attribute__((ext_vector_type(4)))  __bf16 v4bf;
typedef __attribute__((ext_vector_type(8)))  float  v8f;
typedef __attribute__((ext_vector_type(4)))  float  v4f;

// ---- problem dims ----
#define B_    8
#define H_    128
#define W_    128
#define C_    128
#define HP_   130      // padded H (1-px halo)
#define WP_   130      // padded W
#define NWIN_ 256      // 16*16 windows
#define D_    8192     // 8*8*128 patch K-dim
#define K1_   1152     // 3*3*128 conv K-dim
#define OH_   32
#define OW_   32
#define CO_   128
#define AHP_  18       // padded act H (16 + halo for hi in [-1,16])
#define AWP_  18

__device__ __forceinline__ __bf16 f2bf(float f) {
  union { float f; uint32_t u; } v; v.f = f;
  uint32_t r = v.u + 0x7FFFu + ((v.u >> 16) & 1u);   // round-to-nearest-even
  union { uint16_t s; __bf16 b; } o; o.s = (uint16_t)(r >> 16);
  return o.b;
}
__device__ __forceinline__ float bf2f(__bf16 b) {
  union { uint16_t s; __bf16 b; } i; i.b = b;
  union { float f; uint32_t u; } o; o.u = ((uint32_t)i.s) << 16;
  return o.f;
}
__device__ __forceinline__ float leaky(float x) { return x >= 0.f ? x : ALPHA * x; }

__device__ __forceinline__ v16bf make_frag(const __bf16* p0, const __bf16* p1) {
  v8bf a0 = *(const v8bf*)p0;
  v8bf a1 = *(const v8bf*)p1;
  v16bf a;
  #pragma unroll
  for (int j = 0; j < 8; ++j) { a[j] = a0[j]; a[8 + j] = a1[j]; }
  return a;
}

// -------------------- zero fill (for padded buffers) --------------------
__global__ __launch_bounds__(256) void zero_bf16(__bf16* __restrict__ p, int n8) {
  int i = blockIdx.x * blockDim.x + threadIdx.x;
  if (i >= n8) return;
  v8bf z = {};
  *((v8bf*)p + i) = z;
}

// -------------------- prep: fp32 -> bf16 input, written into halo-padded layout ----
__global__ __launch_bounds__(256) void cvt_x_pad(const float* __restrict__ x,
                                                 __bf16* __restrict__ xbp, int n4) {
  int i = blockIdx.x * blockDim.x + threadIdx.x;
  if (i >= n4) return;
  int e   = i * 4;
  int row = e >> 14;          // b*128 + h   (row = 128*128 elems = w*ci)
  int off = e & 16383;
  int b = row >> 7, h = row & 127;
  v4f v = *((const v4f*)x + i);
  v4bf o;
  #pragma unroll
  for (int j = 0; j < 4; ++j) o[j] = f2bf(v[j]);
  __bf16* dst = xbp + ((size_t)(b * HP_ + h + 1) * WP_ + 1) * C_ + off;
  *(v4bf*)dst = o;
}

// transpose weights to [co][k] (k contiguous) bf16
__global__ __launch_bounds__(256) void prep_weights(const float* __restrict__ Wq,
                                                    const float* __restrict__ Wk,
                                                    const float* __restrict__ Wt,
                                                    __bf16* __restrict__ WqT,
                                                    __bf16* __restrict__ WkT,
                                                    __bf16* __restrict__ WtT) {
  int id = blockIdx.x * blockDim.x + threadIdx.x;
  if (id < 147456) {                       // WqT[co*1152+k] = Wq[k*128+co]
    int co = id / K1_, k = id - co * K1_;
    WqT[id] = f2bf(Wq[k * 128 + co]);
  } else if (id < 294912) {
    int id2 = id - 147456;
    int co = id2 / K1_, k = id2 - co * K1_;
    WkT[id2] = f2bf(Wk[k * 128 + co]);
  } else if (id < 425984) {                // WtT[co*4096+t] = Wt[t*128+co]
    int id2 = id - 294912;
    int co = id2 >> 12, t = id2 & 4095;
    WtT[id2] = f2bf(Wt[t * 128 + co]);
  }
}

// -------------------- conv 3x3 SAME as implicit GEMM (bf16 WMMA, halo-padded) -------
// grid (1024, 8): blockIdx.x -> (b,h) row, 8 waves cover w0=0..112; blockIdx.y -> n-tile.
// No bounds checks: input is zero-padded.  36 dense WMMAs per wave.
// Writes pre-BN conv output as bf16 directly in patch layout.
__global__ __launch_bounds__(256)
void conv3x3_wmma(const __bf16* __restrict__ xbp, const __bf16* __restrict__ wT,
                  __bf16* __restrict__ patch) {
  const int tid  = threadIdx.x;
  const int L    = tid & 31;
  const int wave = tid >> 5;
  const int l15  = L & 15;
  const int half = L >> 4;

  const int bh = blockIdx.x;
  const int b  = bh >> 7;
  const int h  = bh & 127;
  const int w0 = wave << 4;
  const int n0 = blockIdx.y << 4;

  const int co = n0 + l15;      // B column / store channel
  const int wm = w0 + l15;      // A row pixel (w)

  const __bf16* wbase = wT + (size_t)co * K1_ + half * 16;

  v8f c = {};
  #pragma unroll 3
  for (int tap = 0; tap < 9; ++tap) {
    const int dh = tap / 3;
    const int dw = tap - dh * 3;
    // padded coords: src row = h+dh (in [0,129]), src col = wm+dw (in [0,129])
    const __bf16* ap = xbp + (((size_t)(b * HP_ + h + dh) * WP_) + (wm + dw)) * C_ + half * 16;
    const __bf16* bp = wbase + tap * 128;
    #pragma unroll
    for (int q = 0; q < 4; ++q) {
      const int ci0 = q << 5;
      v16bf a  = make_frag(ap + ci0, ap + ci0 + 8);
      v16bf bb = make_frag(bp + ci0, bp + ci0 + 8);
      c = __builtin_amdgcn_wmma_f32_16x16x32_bf16(false, a, false, bb,
                                                  (short)0, c, false, false);
    }
  }
  // store: pixel w = w0 + M, M = r + 8*half  ->  win col = w0/8 + half, w%8 = r
  const int win = (h >> 3) * 16 + (w0 >> 3) + half;
  __bf16* op = patch + (((size_t)b * NWIN_ + win) * D_ + (size_t)((h & 7) * 8) * C_ + co);
  #pragma unroll
  for (int r = 0; r < 8; ++r) op[(size_t)r * C_] = f2bf(c[r]);
}

// -------------------- sim = Kp @ Qp^T per batch (256x256, K=8192) --------------------
__global__ __launch_bounds__(256)
void sim_wmma_k(const __bf16* __restrict__ Kc, const __bf16* __restrict__ Qc,
                const float* __restrict__ bias, float* __restrict__ sim) {
  const int tid  = threadIdx.x;
  const int L    = tid & 31;
  const int wave = tid >> 5;
  const int l15  = L & 15;
  const int half = L >> 4;

  const int tile = blockIdx.x * 8 + wave;
  const int b  = tile >> 8;
  const int t  = tile & 255;
  const int f0 = (t >> 4) << 4;
  const int q0 = (t & 15) << 4;

  const __bf16* ap = Kc + ((size_t)(b * NWIN_ + f0 + l15) * D_) + half * 16;
  const __bf16* bp = Qc + ((size_t)(b * NWIN_ + q0 + l15) * D_) + half * 16;

  v8f c = {};
  #pragma unroll 4
  for (int d0 = 0; d0 < D_; d0 += 32) {
    v16bf a  = make_frag(ap + d0, ap + d0 + 8);
    v16bf bb = make_frag(bp + d0, bp + d0 + 8);
    c = __builtin_amdgcn_wmma_f32_16x16x32_bf16(false, a, false, bb,
                                                (short)0, c, false, false);
  }
  const float bqv = bias[q0 + l15];
  float* op = sim + ((size_t)b * NWIN_ * NWIN_ + (size_t)(f0 + 8 * half) * NWIN_ + q0 + l15);
  #pragma unroll
  for (int r = 0; r < 8; ++r) op[(size_t)r * NWIN_] = c[r] + bqv;
}

// -------------------- transposed conv 4x4 stride 2 SAME (implicit GEMM) --------------
// grid 512: blockIdx.x -> (b, ho, wo parity), 8 waves cover co tiles.
// M-tile = 16 outputs with wo = wo0 + 2m.  act is halo-padded (18x18) so the whole
// tap loop is dense: kh parity -> kh0 = ho&1; 4 taps x 8 ci-chunks = 32 WMMAs.
__global__ __launch_bounds__(256)
void tconv_wmma_k(const __bf16* __restrict__ actp, const __bf16* __restrict__ wT,
                  float* __restrict__ T) {
  const int tid  = threadIdx.x;
  const int L    = tid & 31;
  const int wave = tid >> 5;
  const int l15  = L & 15;
  const int half = L >> 4;

  const int mt  = blockIdx.x;
  const int b   = mt >> 6;
  const int r1  = mt & 63;
  const int ho  = r1 >> 1;
  const int wo0 = r1 & 1;
  const int n0  = wave << 4;
  const int co  = n0 + l15;
  const int kh0 = ho & 1;                     // (ho+kh) even  =>  kh = kh0, kh0+2

  v8f c = {};
  #pragma unroll
  for (int u = 0; u < 2; ++u) {
    const int kh  = kh0 + 2 * u;
    const int hi1 = ((ho + kh - 2) >> 1) + 1;             // padded row in [0,17]
    #pragma unroll
    for (int j2 = 0; j2 < 2; ++j2) {
      const int kw  = wo0 + 2 * j2;
      const int wi1 = l15 + wo0 + j2;                     // padded col in [0,17]
      const __bf16* ap = actp + ((size_t)((b * AHP_ + hi1) * AWP_ + wi1) * 256) + half * 16;
      const __bf16* bp = wT + ((size_t)(co * 16 + kh * 4 + kw) * 256) + half * 16;
      #pragma unroll
      for (int q = 0; q < 8; ++q) {
        const int ci0 = q << 5;
        v16bf a  = make_frag(ap + ci0, ap + ci0 + 8);
        v16bf bb = make_frag(bp + ci0, bp + ci0 + 8);
        c = __builtin_amdgcn_wmma_f32_16x16x32_bf16(false, a, false, bb,
                                                    (short)0, c, false, false);
      }
    }
  }
  float* op = T + ((size_t)((b * OH_ + ho) * OW_ + wo0) * CO_ + co);
  #pragma unroll
  for (int r = 0; r < 8; ++r) op[(size_t)(2 * (r + 8 * half)) * CO_] = c[r];
}

// -------------------- deterministic two-stage BN statistics --------------------
__global__ void stats_partial_bf16(const __bf16* __restrict__ data, int rows,
                                   float* __restrict__ partial) {
  const int chans = blockDim.x;
  const int t = threadIdx.x;
  float s1 = 0.f, s2 = 0.f;
  for (int r = blockIdx.x; r < rows; r += gridDim.x) {
    float v = bf2f(data[(size_t)r * chans + t]);
    s1 += v; s2 += v * v;
  }
  partial[(size_t)blockIdx.x * 2 * chans + t] = s1;
  partial[(size_t)blockIdx.x * 2 * chans + chans + t] = s2;
}

__global__ void stats_partial_f32(const float* __restrict__ data, int rows,
                                  float* __restrict__ partial) {
  const int chans = blockDim.x;
  const int t = threadIdx.x;
  float s1 = 0.f, s2 = 0.f;
  for (int r = blockIdx.x; r < rows; r += gridDim.x) {
    float v = data[(size_t)r * chans + t];
    s1 += v; s2 += v * v;
  }
  partial[(size_t)blockIdx.x * 2 * chans + t] = s1;
  partial[(size_t)blockIdx.x * 2 * chans + chans + t] = s2;
}

__global__ void combine_finalize(const float* __restrict__ partial, int nblk,
                                 const float* __restrict__ gamma,
                                 const float* __restrict__ beta,
                                 float invN, float* __restrict__ scale,
                                 float* __restrict__ shift) {
  const int chans = blockDim.x;
  const int t = threadIdx.x;
  float s1 = 0.f, s2 = 0.f;
  for (int k = 0; k < nblk; ++k) {             // fixed order -> deterministic
    s1 += partial[(size_t)k * 2 * chans + t];
    s2 += partial[(size_t)k * 2 * chans + chans + t];
  }
  float mean = s1 * invN;
  float var  = s2 * invN - mean * mean;
  float sc   = gamma[t] * rsqrtf(var + BN_EPS);
  scale[t] = sc;
  shift[t] = beta[t] - mean * sc;
}

// -------------------- elementwise BN + leaky --------------------
__global__ __launch_bounds__(256)
void bnleaky_bf16_inplace(__bf16* __restrict__ d, const float* __restrict__ scale,
                          const float* __restrict__ shift, int total8) {
  int i = blockIdx.x * blockDim.x + threadIdx.x;
  if (i >= total8) return;
  const int c0 = (i * 8) & 127;
  v8bf v = *((v8bf*)d + i);
  #pragma unroll
  for (int j = 0; j < 8; ++j)
    v[j] = f2bf(leaky(scale[c0 + j] * bf2f(v[j]) + shift[c0 + j]));
  *((v8bf*)d + i) = v;
}

// BN+leaky on sim, packed bf16 into halo-padded act layout (18x18 spatial)
__global__ __launch_bounds__(256)
void actpack_pad(const float* __restrict__ sim, const float* __restrict__ scale,
                 const float* __restrict__ shift, __bf16* __restrict__ actp, int total8) {
  int i = blockIdx.x * blockDim.x + threadIdx.x;
  if (i >= total8) return;
  const int c0 = (i * 8) & 255;
  const int p  = i >> 5;                 // pixel index: b*256 + hi*16 + wi
  const int b  = p >> 8;
  const int hi = (p >> 4) & 15;
  const int wi = p & 15;
  const float* src = sim + (size_t)i * 8;
  v8bf o;
  #pragma unroll
  for (int j = 0; j < 8; ++j)
    o[j] = f2bf(leaky(scale[c0 + j] * src[j] + shift[c0 + j]));
  __bf16* dst = actp + ((size_t)((b * AHP_ + hi + 1) * AWP_ + wi + 1) * 256) + c0;
  *(v8bf*)dst = o;
}

__global__ __launch_bounds__(256)
void outapply(const float* __restrict__ T, const float* __restrict__ scale,
              const float* __restrict__ shift, float* __restrict__ out, int total4) {
  int i = blockIdx.x * blockDim.x + threadIdx.x;
  if (i >= total4) return;
  const int c0 = (i * 4) & 127;
  v4f v = *((const v4f*)T + i);
  #pragma unroll
  for (int j = 0; j < 4; ++j) v[j] = leaky(scale[c0 + j] * v[j] + shift[c0 + j]);
  *((v4f*)out + i) = v;
}

// -------------------- host orchestration --------------------
extern "C" void kernel_launch(void* const* d_in, const int* in_sizes, int n_in,
                              void* d_out, int out_size, void* d_ws, size_t ws_size,
                              hipStream_t stream) {
  const float* x    = (const float*)d_in[0];
  const float* Wq   = (const float*)d_in[1];
  const float* gq   = (const float*)d_in[2];
  const float* bq   = (const float*)d_in[3];
  const float* Wk   = (const float*)d_in[4];
  const float* gk   = (const float*)d_in[5];
  const float* bk   = (const float*)d_in[6];
  const float* bias = (const float*)d_in[7];
  const float* g1   = (const float*)d_in[8];
  const float* b1   = (const float*)d_in[9];
  const float* Wt   = (const float*)d_in[10];
  const float* gt   = (const float*)d_in[11];
  const float* bt   = (const float*)d_in[12];
  float* out = (float*)d_out;
  (void)in_sizes; (void)n_in; (void)out_size; (void)ws_size;

  char* ws = (char*)d_ws;
  __bf16* xbp = (__bf16*)(ws + 0);            // 8*130*130*128*2 = 34,611,200 B
  __bf16* WqT = (__bf16*)(ws + 34611200);     //    294,912 B
  __bf16* WkT = (__bf16*)(ws + 34906112);     //    294,912 B
  __bf16* WtT = (__bf16*)(ws + 35201024);     //    262,144 B
  __bf16* Qc  = (__bf16*)(ws + 35463168);     // 33,554,432 B (patch layout, bf16)
  __bf16* Kc  = (__bf16*)(ws + 69017600);     // 33,554,432 B
  float*  sim = (float*) (ws + 102572032);    //  2,097,152 B
  __bf16* actp= (__bf16*)(ws + 104669184);    //  8*18*18*256*2 = 1,327,104 B
  float*  T   = (float*) (ws + 105996288);    //  4,194,304 B
  float*  st  = (float*) (ws + 110190592);    // scales/shifts + partials

  float* scQ = st;         float* shQ = st + 128;
  float* scK = st + 256;   float* shK = st + 384;
  float* scS = st + 512;   float* shS = st + 768;
  float* scO = st + 1024;  float* shO = st + 1152;
  float* part = st + 1280;                    // up to 65,536 floats, reused serially

  // 0) zero the halo-padded buffers (cheap vs. compute)
  zero_bf16<<<8450, 256, 0, stream>>>(xbp, 2163200);   // 17,305,600 elems / 8
  zero_bf16<<<324, 256, 0, stream>>>(actp, 82944);     //    663,552 elems / 8

  // 1) precision conversion + weight transposes
  cvt_x_pad<<<16384, 256, 0, stream>>>(x, xbp, 4194304);
  prep_weights<<<1664, 256, 0, stream>>>(Wq, Wk, Wt, WqT, WkT, WtT);

  // 2) Q,K convs (WMMA), written pre-BN as bf16 in patch layout
  conv3x3_wmma<<<dim3(1024, 8), 256, 0, stream>>>(xbp, WqT, Qc);
  conv3x3_wmma<<<dim3(1024, 8), 256, 0, stream>>>(xbp, WkT, Kc);

  // 3) BN stats (deterministic two-stage), then BN+leaky in place
  stats_partial_bf16<<<256, 128, 0, stream>>>(Qc, 131072, part);
  combine_finalize<<<1, 128, 0, stream>>>(part, 256, gq, bq, 1.0f / 131072.0f, scQ, shQ);
  stats_partial_bf16<<<256, 128, 0, stream>>>(Kc, 131072, part);
  combine_finalize<<<1, 128, 0, stream>>>(part, 256, gk, bk, 1.0f / 131072.0f, scK, shK);
  bnleaky_bf16_inplace<<<8192, 256, 0, stream>>>(Qc, scQ, shQ, 2097152);
  bnleaky_bf16_inplace<<<8192, 256, 0, stream>>>(Kc, scK, shK, 2097152);

  // 4) similarity GEMM (WMMA) + beta_bias
  sim_wmma_k<<<256, 256, 0, stream>>>(Kc, Qc, bias, sim);

  // 5) BN over (b, f) per q-channel, leaky, pack bf16 into padded act
  stats_partial_f32<<<32, 256, 0, stream>>>(sim, 2048, part);
  combine_finalize<<<1, 256, 0, stream>>>(part, 32, g1, b1, 1.0f / 2048.0f, scS, shS);
  actpack_pad<<<256, 256, 0, stream>>>(sim, scS, shS, actp, 65536);

  // 6) transposed conv (WMMA), final BN + leaky
  tconv_wmma_k<<<512, 256, 0, stream>>>(actp, WtT, T);
  stats_partial_f32<<<128, 128, 0, stream>>>(T, 8192, part);
  combine_finalize<<<1, 128, 0, stream>>>(part, 128, gt, bt, 1.0f / 8192.0f, scO, shO);
  outapply<<<1024, 256, 0, stream>>>(T, scO, shO, out, 262144);
}
